// MultiHeadAttention_54314156425628
// MI455X (gfx1250) — matrix operand order repaired and checked by reading, not compiled, not run on hardware
//
#include <hip/hip_runtime.h>

// ---------------------------------------------------------------------------
// MultiHeadAttention for MI455X (gfx1250, wave32, WMMA 16x16x32 f16 + TDM)
//   b=2, s=2048, e=1024, h=16, d=64
//   Pipeline: 3x projection GEMM (f32 in -> f16 out, f32 accum via WMMA)
//             + flash attention per (b,h) head slice (contiguous by reshape),
//               K/V tiles staged by the Tensor Data Mover, double-buffered.
// ---------------------------------------------------------------------------

#define BATCH 2
#define SEQ   2048
#define EMB   1024
#define HEADS 16
#define DHEAD 64

typedef _Float16 v16h __attribute__((ext_vector_type(16)));
typedef float    v8f  __attribute__((ext_vector_type(8)));

typedef unsigned int tdm_u32x4 __attribute__((ext_vector_type(4)));
typedef int          tdm_i32x8 __attribute__((ext_vector_type(8)));
typedef int          tdm_i32x4 __attribute__((ext_vector_type(4)));

#if defined(__has_builtin)
#if __has_builtin(__builtin_amdgcn_tensor_load_to_lds)
#define USE_TDM 1
#endif
#endif
#ifndef USE_TDM
#define USE_TDM 0
#endif

union Frag {
    v16h h;
    uint4 q[2];
    _Float16 e[16];
};

__device__ __forceinline__ v8f wmma16(v16h a, v16h b, v8f c) {
    // D = A(16x32 f16) * B(32x16 f16) + C(16x16 f32)
    return __builtin_amdgcn_wmma_f32_16x16x32_f16(
        false, a, false, b, (short)0, c, false, false);
}

__device__ __forceinline__ v8f zero8() {
    v8f z;
#pragma unroll
    for (int i = 0; i < 8; ++i) z[i] = 0.0f;
    return z;
}

#if USE_TDM
__device__ __forceinline__ unsigned lds_addr_of(const void* p) {
    // generic -> LDS address space -> raw 32-bit LDS byte offset
    return (unsigned)(unsigned long long)
        (__attribute__((address_space(3))) const void*)p;
}

// 2D TDM load: tile_dim0 x tile_dim1 elements of 2 bytes, row stride
// stride0 (elements) in global memory; LDS rows padded +16B per 128B row
// (pad_interval=4 -> every 32 DWORDs, pad_amount=3 -> +4 DWORDs) so the
// LDS layout is rows of 72 halfs.
__device__ __forceinline__ void tdm_load_2d(unsigned lds_byte,
                                            const _Float16* gsrc,
                                            unsigned tile_d0, unsigned tile_d1,
                                            unsigned tensor_d0, unsigned tensor_d1,
                                            unsigned stride0)
{
    const unsigned long long ga = (unsigned long long)gsrc;
    tdm_u32x4 g0;
    g0[0] = 1u;                                        // count=1, user descriptor
    g0[1] = lds_byte;                                  // lds_addr
    g0[2] = (unsigned)(ga & 0xFFFFFFFFu);              // global_addr[31:0]
    g0[3] = ((unsigned)(ga >> 32) & 0x01FFFFFFu)       // global_addr[56:32]
            | (2u << 30);                              // type = 2 ("image")

    tdm_i32x8 g1;
    g1[0] = (int)((1u << 16)        // data_size = 1 -> 2 bytes
                | (1u << 20)        // pad_enable
                | (4u << 22)        // pad_interval: 2^(4+1)=32 DWORDs (128B)
                | (3u << 25));      // pad_amount:   3+1 = 4 DWORDs (16B)
    g1[1] = (int)((tensor_d0 & 0xFFFFu) << 16);                        // dim0[15:0]
    g1[2] = (int)(((tensor_d0 >> 16) & 0xFFFFu)
                | ((tensor_d1 & 0xFFFFu) << 16));                      // dim0 hi | dim1 lo
    g1[3] = (int)(((tensor_d1 >> 16) & 0xFFFFu) | (tile_d0 << 16));    // dim1 hi | tile0
    g1[4] = (int)tile_d1;                                              // tile1 | tile2=0
    g1[5] = (int)stride0;                                              // dim0_stride lo
    g1[6] = 0;                                                         // stride hi, dim1_stride
    g1[7] = 0;

    tdm_i32x4 z4;
    z4[0] = 0; z4[1] = 0; z4[2] = 0; z4[3] = 0;        // groups 2/3: 2D tensor
    tdm_i32x8 z8;
#pragma unroll
    for (int i = 0; i < 8; ++i) z8[i] = 0;

    // 6-arg form (this toolchain): (g0, g1, g2, g3, extra, cpol)
    __builtin_amdgcn_tensor_load_to_lds(g0, g1, z4, z4, z8, 0);
}

__device__ __forceinline__ void wait_tensor0() {
#if __has_builtin(__builtin_amdgcn_s_wait_tensorcnt)
    __builtin_amdgcn_s_wait_tensorcnt(0);
#else
    asm volatile("s_wait_tensorcnt 0" ::: "memory");
#endif
}
#endif // USE_TDM

// ---------------------------------------------------------------------------
// Projection GEMM: C[4096,1024] = X[4096,1024] @ W[1024,1024]^T + bias
//   vmode 0: store f16 row-major [m][n]  (Q and K projections)
//   vmode 1: store f16 as per-head transposed V: Vt[(b*16+h)*64 + d][s]
// Tile: 128x128x32, 256 threads (8 waves), wave = 64(M) x 32(N)
// ---------------------------------------------------------------------------
__global__ __launch_bounds__(256)
void proj_gemm_kernel(const float* __restrict__ X,
                      const float* __restrict__ W,
                      const float* __restrict__ bias,
                      _Float16* __restrict__ Out,
                      int vmode)
{
    constexpr int BM = 128, BN = 128, BK = 32, LDK = 40; // LDK padded (80B rows)
    __shared__ _Float16 As[BM * LDK];
    __shared__ _Float16 Bs[BN * LDK];

    const int tid  = threadIdx.x;
    const int lane = tid & 31;
    const int wave = tid >> 5;
    const int lm   = lane & 15;
    const int lh   = lane >> 4;

    const int bm = blockIdx.y * BM;
    const int bn = blockIdx.x * BN;
    const int wm = (wave >> 2) * 64;   // 0 / 64
    const int wn = (wave & 3) * 32;    // 0 / 32 / 64 / 96

    v8f acc[4][2];
#pragma unroll
    for (int i = 0; i < 4; ++i)
#pragma unroll
        for (int j = 0; j < 2; ++j) acc[i][j] = zero8();

    for (int k0 = 0; k0 < EMB; k0 += BK) {
        __syncthreads();
        // ---- fill LDS tiles (f32 -> f16 convert on the fly) ----
#pragma unroll
        for (int r = 0; r < 4; ++r) {
            const int eidx = (tid + r * 256) * 4;        // 0..4095
            const int row  = eidx >> 5;                  // /32
            const int col  = eidx & 31;
            union { _Float16 h[4]; uint2 u; } cv;

            const float4 xa = *reinterpret_cast<const float4*>(
                &X[(size_t)(bm + row) * EMB + k0 + col]);
            cv.h[0] = (_Float16)xa.x; cv.h[1] = (_Float16)xa.y;
            cv.h[2] = (_Float16)xa.z; cv.h[3] = (_Float16)xa.w;
            *reinterpret_cast<uint2*>(&As[row * LDK + col]) = cv.u;

            const float4 wa = *reinterpret_cast<const float4*>(
                &W[(size_t)(bn + row) * EMB + k0 + col]);
            cv.h[0] = (_Float16)wa.x; cv.h[1] = (_Float16)wa.y;
            cv.h[2] = (_Float16)wa.z; cv.h[3] = (_Float16)wa.w;
            *reinterpret_cast<uint2*>(&Bs[row * LDK + col]) = cv.u;
        }
        __syncthreads();

        // ---- fragments + WMMA ----
        Frag bf[2];
#pragma unroll
        for (int j = 0; j < 2; ++j) {
            const _Float16* p = &Bs[(wn + j * 16 + lm) * LDK + lh * 16];
            bf[j].q[0] = *reinterpret_cast<const uint4*>(p);
            bf[j].q[1] = *reinterpret_cast<const uint4*>(p + 8);
        }
#pragma unroll
        for (int i = 0; i < 4; ++i) {
            Frag af;
            const _Float16* p = &As[(wm + i * 16 + lm) * LDK];
            af.q[0] = *reinterpret_cast<const uint4*>(p + lh * 16);
            af.q[1] = *reinterpret_cast<const uint4*>(p + lh * 16 + 8);
#pragma unroll
            for (int j = 0; j < 2; ++j)
                acc[i][j] = wmma16(af.h, bf[j].h, acc[i][j]);
        }
    }

    // ---- epilogue: bias add, convert, store ----
#pragma unroll
    for (int j = 0; j < 2; ++j) {
        const int n = bn + wn + j * 16 + lm;     // column (fixed per lane)
        const float bv = bias[n];
#pragma unroll
        for (int i = 0; i < 4; ++i) {
#pragma unroll
            for (int v = 0; v < 8; ++v) {
                const int m = bm + wm + i * 16 + lh * 8 + v;
                const float val = acc[i][j][v] + bv;
                if (vmode == 0) {
                    Out[(size_t)m * EMB + n] = (_Float16)val;
                } else {
                    // map (m,n) -> (b,h,s,d) of the bug-faithful reshape,
                    // store V transposed per head: Vt[(b*16+h)*64 + d][s]
                    const int b_i = m >> 11;             // m / 2048
                    const int r   = m & 2047;
                    const int h_i = r >> 7;              // r / 128
                    const int s_i = ((r & 127) << 4) + (n >> 6);
                    const int d_i = n & 63;
                    Out[(((size_t)(b_i * HEADS + h_i) * DHEAD + d_i) << 11) + s_i]
                        = (_Float16)val;
                }
            }
        }
    }
}

// ---------------------------------------------------------------------------
// Flash attention per (b,h): Q,K contiguous [2048,64] f16, Vt [64,2048] f16.
// Block: 128 q rows (8 waves x 16 rows). 64-key tiles staged by TDM into
// double-buffered LDS (72-half padded rows via hardware padding). Causal +
// pad mask, bug-faithful scale (* sqrt(64) = 8). Output f32, contiguous.
// ---------------------------------------------------------------------------
__global__ __launch_bounds__(256)
void attn_kernel(const _Float16* __restrict__ Qp,
                 const _Float16* __restrict__ Kp,
                 const _Float16* __restrict__ Vt,
                 const unsigned char* __restrict__ padmask,
                 float* __restrict__ Out)
{
    constexpr int BQ = 128, BKEY = 64;
    constexpr int LKV = 72;   // padded row stride (144B) == TDM pad config
    constexpr int LP  = 72;
    constexpr float LOG2E = 1.4426950408889634f;

    __shared__ _Float16 Ks[2][BKEY * LKV];     // K tile  [key][d], double buffer
    __shared__ _Float16 Vs[2][DHEAD * LKV];    // Vt tile [d][key], double buffer
    __shared__ _Float16 Ps[8 * 16 * LP];       // per-wave P staging [16][LP]

    const int tid  = threadIdx.x;
    const int lane = tid & 31;
    const int wave = tid >> 5;
    const int lm   = lane & 15;
    const int lh   = lane >> 4;

    const int bh    = blockIdx.y;              // b*16 + h
    const int b_i   = bh >> 4;
    const int qbase = blockIdx.x * BQ;
    const size_t headOff = (size_t)bh * SEQ * DHEAD;

    const _Float16* Qh = Qp + headOff;
    const _Float16* Kh = Kp + headOff;
    const _Float16* Vh = Vt + headOff;                    // [d][s]
    const unsigned char* mask = padmask + (size_t)b_i * SEQ;

    // --- Q fragments (A-layout), loaded once from global ---
    const int qrow = qbase + wave * 16 + lm;
    Frag qf[2];
#pragma unroll
    for (int c = 0; c < 2; ++c) {
        const _Float16* p = &Qh[(size_t)qrow * DHEAD + c * 32];
        qf[c].q[0] = *reinterpret_cast<const uint4*>(p + lh * 16);
        qf[c].q[1] = *reinterpret_cast<const uint4*>(p + lh * 16 + 8);
    }

    v8f o[4];
#pragma unroll
    for (int t = 0; t < 4; ++t) o[t] = zero8();
    float mrow[8], lrow[8];
#pragma unroll
    for (int v = 0; v < 8; ++v) { mrow[v] = -1e30f; lrow[v] = 0.0f; }

    _Float16* Pw = &Ps[wave * 16 * LP];
    const int kend = qbase + BQ;               // causal block bound

#if USE_TDM
    // TDM issues regardless of EXEC -> must be behind a *scalar* branch so
    // only wave 0 issues the DMA. readfirstlane makes the condition uniform.
    const bool issuer = (__builtin_amdgcn_readfirstlane(wave) == 0);
    if (issuer) {
        tdm_load_2d(lds_addr_of(&Ks[0][0]), Kh,      BKEY, BKEY, DHEAD, SEQ,  DHEAD);
        tdm_load_2d(lds_addr_of(&Vs[0][0]), Vh,      BKEY, DHEAD, SEQ,  DHEAD, SEQ);
    }
#endif

    int ibuf = 0;
    for (int kb = 0; kb < kend; kb += BKEY, ibuf ^= 1) {
#if USE_TDM
        if (issuer) wait_tensor0();            // tile ibuf landed in LDS
        __syncthreads();                       // publish; everyone past buf^1
        if (kb + BKEY < kend) {
            if (issuer) {                      // DMA next tile during compute
                tdm_load_2d(lds_addr_of(&Ks[ibuf ^ 1][0]),
                            Kh + (size_t)(kb + BKEY) * DHEAD,
                            BKEY, BKEY, DHEAD, SEQ, DHEAD);
                tdm_load_2d(lds_addr_of(&Vs[ibuf ^ 1][0]),
                            Vh + (kb + BKEY),
                            BKEY, DHEAD, SEQ, DHEAD, SEQ);
            }
        }
#else
        __syncthreads();
        {   // synchronous fallback: 16 halfs / thread for each of K and Vt
            const int row = tid >> 2;                    // 0..63
            const int col = (tid & 3) * 16;              // 0/16/32/48
            const uint4* gk = reinterpret_cast<const uint4*>(
                &Kh[(size_t)(kb + row) * DHEAD + col]);
            uint4 a0 = gk[0], a1 = gk[1];
            *reinterpret_cast<uint4*>(&Ks[ibuf][row * LKV + col])     = a0;
            *reinterpret_cast<uint4*>(&Ks[ibuf][row * LKV + col + 8]) = a1;

            const uint4* gv = reinterpret_cast<const uint4*>(
                &Vh[(size_t)row * SEQ + kb + col]);
            uint4 v0 = gv[0], v1 = gv[1];
            *reinterpret_cast<uint4*>(&Vs[ibuf][row * LKV + col])     = v0;
            *reinterpret_cast<uint4*>(&Vs[ibuf][row * LKV + col + 8]) = v1;

            if (kb + BKEY < kend) {
                __builtin_prefetch(&Kh[(size_t)(kb + BKEY + row) * DHEAD + col], 0, 2);
                __builtin_prefetch(&Vh[(size_t)row * SEQ + kb + BKEY + col], 0, 2);
            }
        }
        __syncthreads();
#endif
        const _Float16* Ktile = &Ks[ibuf][0];
        const _Float16* Vtile = &Vs[ibuf][0];

        // ---- S = Q @ K^T  (4 key tiles of 16, accumulate d in 2 chunks) ----
        v8f s[4];
#pragma unroll
        for (int j = 0; j < 4; ++j) {
            s[j] = zero8();
#pragma unroll
            for (int c = 0; c < 2; ++c) {
                Frag kf;
                const _Float16* p = &Ktile[(j * 16 + lm) * LKV + c * 32 + lh * 16];
                kf.q[0] = *reinterpret_cast<const uint4*>(p);
                kf.q[1] = *reinterpret_cast<const uint4*>(p + 8);
                s[j] = wmma16(qf[c].h, kf.h, s[j]);
            }
        }

        // ---- scale (bug-faithful *8), pad + causal mask ----
#pragma unroll
        for (int j = 0; j < 4; ++j) {
            const int kcol = kb + j * 16 + lm;
            const bool pm = (mask[kcol] != 0);
#pragma unroll
            for (int v = 0; v < 8; ++v) {
                const int qg = qbase + wave * 16 + lh * 8 + v;
                float val = s[j][v] * 8.0f;
                if (pm || (kcol > qg)) val = -1e30f;
                s[j][v] = val;
            }
        }

        // ---- online softmax update ----
        float fac[8];
#pragma unroll
        for (int v = 0; v < 8; ++v) {
            float rm = fmaxf(fmaxf(s[0][v], s[1][v]), fmaxf(s[2][v], s[3][v]));
#pragma unroll
            for (int off = 1; off < 16; off <<= 1)
                rm = fmaxf(rm, __shfl_xor(rm, off, 32));
            const float nm = fmaxf(mrow[v], rm);
            fac[v] = exp2f((mrow[v] - nm) * LOG2E);
            mrow[v] = nm;
        }
#pragma unroll
        for (int j = 0; j < 4; ++j)
#pragma unroll
            for (int v = 0; v < 8; ++v)
                s[j][v] = exp2f((s[j][v] - mrow[v]) * LOG2E);
#pragma unroll
        for (int v = 0; v < 8; ++v) {
            float rs = (s[0][v] + s[1][v]) + (s[2][v] + s[3][v]);
#pragma unroll
            for (int off = 1; off < 16; off <<= 1)
                rs += __shfl_xor(rs, off, 32);
            lrow[v] = lrow[v] * fac[v] + rs;
#pragma unroll
            for (int t = 0; t < 4; ++t) o[t][v] *= fac[v];
        }

        // ---- P (C-layout f32) -> per-wave LDS -> A-layout f16 ----
#pragma unroll
        for (int j = 0; j < 4; ++j)
#pragma unroll
            for (int v = 0; v < 8; ++v)
                Pw[(lh * 8 + v) * LP + j * 16 + lm] = (_Float16)s[j][v];
        asm volatile("s_wait_dscnt 0" ::: "memory");   // same-wave DS RAW fence

        // ---- O += P @ V ----
#pragma unroll
        for (int c = 0; c < 2; ++c) {                  // 2 chunks of 32 keys
            Frag pf;
            const _Float16* p = &Pw[lm * LP + c * 32];
            pf.q[0] = *reinterpret_cast<const uint4*>(p + lh * 16);
            pf.q[1] = *reinterpret_cast<const uint4*>(p + lh * 16 + 8);
#pragma unroll
            for (int t = 0; t < 4; ++t) {              // 4 d-tiles of 16
                Frag vf;
                const _Float16* pv = &Vtile[(t * 16 + lm) * LKV + c * 32 + lh * 16];
                vf.q[0] = *reinterpret_cast<const uint4*>(pv);
                vf.q[1] = *reinterpret_cast<const uint4*>(pv + 8);
                o[t] = wmma16(pf.h, vf.h, o[t]);
            }
        }
    }

    // ---- epilogue: normalize and store f32 (contiguous = final reshape) ----
    float* Oh = Out + headOff;
#pragma unroll
    for (int t = 0; t < 4; ++t) {
#pragma unroll
        for (int v = 0; v < 8; ++v) {
            const int qg = qbase + wave * 16 + lh * 8 + v;
            const int d  = t * 16 + lm;
            Oh[(size_t)qg * DHEAD + d] = o[t][v] / lrow[v];
        }
    }
}

// ---------------------------------------------------------------------------
extern "C" void kernel_launch(void* const* d_in, const int* in_sizes, int n_in,
                              void* d_out, int out_size, void* d_ws, size_t ws_size,
                              hipStream_t stream) {
    const float* q  = (const float*)d_in[0];
    const float* k  = (const float*)d_in[1];
    const float* v  = (const float*)d_in[2];
    const unsigned char* mask = (const unsigned char*)d_in[3];  // bool [b,s]
    const float* Wq = (const float*)d_in[4];
    const float* bq = (const float*)d_in[5];
    const float* Wk = (const float*)d_in[6];
    const float* bk = (const float*)d_in[7];
    const float* Wv = (const float*)d_in[8];
    const float* bv = (const float*)d_in[9];

    const size_t elems = (size_t)BATCH * SEQ * EMB;   // 4M f16 each
    _Float16* Qp = (_Float16*)d_ws;
    _Float16* Kp = Qp + elems;
    _Float16* Vt = Kp + elems;

    dim3 pgrid(EMB / 128, (BATCH * SEQ) / 128);       // 8 x 32
    proj_gemm_kernel<<<pgrid, 256, 0, stream>>>(q, Wq, bq, Qp, 0);
    proj_gemm_kernel<<<pgrid, 256, 0, stream>>>(k, Wk, bk, Kp, 0);
    proj_gemm_kernel<<<pgrid, 256, 0, stream>>>(v, Wv, bv, Vt, 1);

    dim3 agrid(SEQ / 128, BATCH * HEADS);             // 16 x 32
    attn_kernel<<<agrid, 256, 0, stream>>>(Qp, Kp, Vt, mask, (float*)d_out);
}
